// ModulationConvolution_35107062677748
// MI455X (gfx1250) — matrix operand order repaired and checked by reading, not compiled, not run on hardware
//
#include <hip/hip_runtime.h>
#include <hip/hip_bf16.h>

// ---------------- CDNA5 WMMA types ----------------
typedef __attribute__((ext_vector_type(16))) __bf16 v16bf;
typedef __attribute__((ext_vector_type(8)))  float  v8f;

union Frag {
    uint4 u[2];
    v16bf v;
};

// bf16 round-to-nearest-even split helpers
__device__ __forceinline__ unsigned short f2bf(float f) {
    unsigned int u = __float_as_uint(f);
    unsigned int r = u + 0x7FFFu + ((u >> 16) & 1u);
    return (unsigned short)(r >> 16);
}
__device__ __forceinline__ float bf2f(unsigned short h) {
    return __uint_as_float(((unsigned int)h) << 16);
}

#define NB   8
#define NQ   16
#define QD   256
#define CIN  256
#define HW   128
#define NTAP 9
#define NO   (CIN * NTAP)          // 2304
#define NKC  8                     // 256 channels / 32 per WMMA K-chunk

// ---------------- Stage 1a: dynamic kernels  Kf[b*16+q][o] = query·W + bias ----
__global__ void __launch_bounds__(256)
gen_kernels(const float* __restrict__ query, const float* __restrict__ W,
            const float* __restrict__ b_lin, float* __restrict__ Kf) {
    int idx = blockIdx.x * 256 + threadIdx.x;          // NB*NQ*NO = 294912
    if (idx >= NB * NQ * NO) return;
    int o  = idx % NO;
    int bq = idx / NO;
    const float* qr = query + (size_t)bq * QD;
    const float* wr = W     + (size_t)o  * QD;
    float acc = b_lin[o];
    #pragma unroll 4
    for (int k = 0; k < QD; k += 4) {
        float4 a = *(const float4*)(qr + k);
        float4 w = *(const float4*)(wr + k);
        acc += a.x * w.x + a.y * w.y + a.z * w.z + a.w * w.w;
    }
    Kf[idx] = acc;
}

// ---------------- Stage 1b: pack A-fragments, 16x32 bf16, contiguous k order ----------
// Frag element e, lane-group g (lane>>4):  K = g*16 + e
// Layout: Ah/Al[ ((((b*9+t)*8+kc)*32)+lane)*16 + e ]
__global__ void __launch_bounds__(256)
pack_afrag(const float* __restrict__ Kf, unsigned short* __restrict__ Ah,
           unsigned short* __restrict__ Al) {
    int idx = blockIdx.x * 256 + threadIdx.x;          // NB*NTAP*NKC*32 = 18432
    if (idx >= NB * NTAP * NKC * 32) return;
    int lane = idx & 31;
    int kc   = (idx >> 5) & 7;
    int bt   = idx >> 8;                               // b*9 + t
    int t    = bt % NTAP;
    int b    = bt / NTAP;
    int m = lane & 15, g = lane >> 4;
    size_t outBase = (size_t)idx * 16;
    const float* krow = Kf + ((size_t)(b * NQ + m)) * NO;
    #pragma unroll
    for (int e = 0; e < 16; ++e) {
        int K = g * 16 + e;
        int c = kc * 32 + K;
        float v = krow[c * NTAP + t];                  // o = c*9 + tap
        unsigned short hi = f2bf(v);
        unsigned short lo = f2bf(v - bf2f(hi));
        Ah[outBase + e] = hi;
        Al[outBase + e] = lo;
    }
}

// ---------------- Stage 2: implicit-GEMM dynamic conv via bf16x3 WMMA ----------
// Block = 256 threads (8 wave32). blockIdx.x = y row, blockIdx.y = batch.
// Wave w owns output tile out[b, 0..15, y, w*16 .. w*16+15].
#define SXW 132   // 128 + halo(2) + pad; keeps [x][ch16] slices 32B aligned

__global__ void __launch_bounds__(256)
modconv(const float* __restrict__ image, const unsigned short* __restrict__ Ah,
        const unsigned short* __restrict__ Al, float* __restrict__ out) {
    __shared__ __align__(32) unsigned short sHi[3][SXW][32];
    __shared__ __align__(32) unsigned short sLo[3][SXW][32];

    const int y    = blockIdx.x;
    const int b    = blockIdx.y;
    const int tid  = threadIdx.x;
    const int wave = tid >> 5;
    const int lane = tid & 31;
    const int n    = lane & 15;          // B column / output x within tile
    const int g    = lane >> 4;          // lane group
    const int x0   = wave * 16;

    v8f acc = {0.f, 0.f, 0.f, 0.f, 0.f, 0.f, 0.f, 0.f};

    for (int kc = 0; kc < NKC; ++kc) {
        const int c0 = kc * 32;
        __syncthreads();   // protect LDS WAR vs previous chunk's readers

        // --- cooperative fill: rows y-1..y+1, channels c0..c0+31, x -1..130 ---
        for (int i = tid; i < 3 * 32 * SXW; i += 256) {
            int r   = i / (32 * SXW);
            int rem = i - r * (32 * SXW);
            int ch  = rem / SXW;
            int sx  = rem - ch * SXW;                  // fastest: coalesced x
            int yy = y + r - 1;
            int xx = sx - 1;
            float v = 0.f;
            if ((unsigned)yy < (unsigned)HW && (unsigned)xx < (unsigned)HW) {
                const float* p = image +
                    (((size_t)(b * CIN + c0 + ch) * HW + yy) * HW + xx);
                v = *p;
                if (kc < NKC - 1)                       // global_prefetch_b8
                    __builtin_prefetch(p + (size_t)32 * HW * HW, 0, 1);
            }
            unsigned short hi = f2bf(v);
            unsigned short lo = f2bf(v - bf2f(hi));
            sHi[0][0][(size_t)(r * SXW + sx) * 32 + ch] = hi;
            sLo[0][0][(size_t)(r * SXW + sx) * 32 + ch] = lo;
        }
        __syncthreads();

        // --- 9 taps: D += A_tap(16x32) x B_tap(32x16), bf16x3 ---
        #pragma unroll
        for (int t = 0; t < NTAP; ++t) {
            const int dy = t / 3;        // LDS row index (y-1 .. y+1)
            const int dx = t % 3 - 1;

            size_t aOff = ((((size_t)(b * NTAP + t) * NKC + kc) * 32) + lane) * 16;
            const uint4* pah = (const uint4*)(Ah + aOff);
            const uint4* pal = (const uint4*)(Al + aOff);
            Frag aH, aL;
            aH.u[0] = pah[0]; aH.u[1] = pah[1];
            aL.u[0] = pal[0]; aL.u[1] = pal[1];

            // B fragment: lane holds 16 consecutive channels at one pixel
            const unsigned short* pbh = &sHi[dy][1 + x0 + n + dx][g * 16];
            const unsigned short* pbl = &sLo[dy][1 + x0 + n + dx][g * 16];
            Frag bH, bL;
            bH.u[0] = ((const uint4*)pbh)[0]; bH.u[1] = ((const uint4*)pbh)[1];
            bL.u[0] = ((const uint4*)pbl)[0]; bL.u[1] = ((const uint4*)pbl)[1];

            acc = __builtin_amdgcn_wmma_f32_16x16x32_bf16(
                false, aH.v, false, bH.v, (short)0, acc, false, false);
            acc = __builtin_amdgcn_wmma_f32_16x16x32_bf16(
                false, aH.v, false, bL.v, (short)0, acc, false, false);
            acc = __builtin_amdgcn_wmma_f32_16x16x32_bf16(
                false, aL.v, false, bH.v, (short)0, acc, false, false);
        }
    }

    // --- store D: lane(l), vgpr(v) -> m = g*8+v, n = lane&15 (ISA C/D layout) ---
    #pragma unroll
    for (int v = 0; v < 8; ++v) {
        int m = g * 8 + v;
        out[(((size_t)(b * NQ + m)) * HW + y) * HW + x0 + n] = acc[v];
    }
}

// ---------------- host entry ----------------
extern "C" void kernel_launch(void* const* d_in, const int* in_sizes, int n_in,
                              void* d_out, int out_size, void* d_ws, size_t ws_size,
                              hipStream_t stream) {
    const float* query = (const float*)d_in[0];   // (8,16,256)
    const float* image = (const float*)d_in[1];   // (8,256,128,128)
    const float* W     = (const float*)d_in[2];   // (2304,256)
    const float* b_lin = (const float*)d_in[3];   // (2304,)
    float* out = (float*)d_out;                   // (8,16,128,128)

    char* ws = (char*)d_ws;
    const size_t KF_BYTES = (size_t)NB * NQ * NO * sizeof(float);          // 1.18 MB
    const size_t AF_BYTES = (size_t)NB * NTAP * NKC * 32 * 16 * 2;         // 0.59 MB
    float*          Kf = (float*)ws;
    unsigned short* Ah = (unsigned short*)(ws + KF_BYTES);
    unsigned short* Al = (unsigned short*)(ws + KF_BYTES + AF_BYTES);
    (void)ws_size; (void)in_sizes; (void)n_in; (void)out_size;

    const int nK = NB * NQ * NO;                  // 294912
    gen_kernels<<<(nK + 255) / 256, 256, 0, stream>>>(query, W, b_lin, Kf);

    const int nF = NB * NTAP * NKC * 32;          // 18432
    pack_afrag<<<(nF + 255) / 256, 256, 0, stream>>>(Kf, Ah, Al);

    dim3 grid(HW, NB);                            // 128 rows x 8 batches
    modconv<<<grid, 256, 0, stream>>>(image, Ah, Al, out);
}
